// Vanilla_25323127177740
// MI455X (gfx1250) — matrix operand order repaired and checked by reading, not compiled, not run on hardware
//
#include <hip/hip_runtime.h>
#include <hip/hip_bf16.h>
#include <math.h>

#define DEV __device__ __forceinline__

typedef __attribute__((ext_vector_type(16))) __bf16 bf16x16;
typedef __attribute__((ext_vector_type(8)))  __bf16 bf16x8;
typedef __attribute__((ext_vector_type(8)))  float  f32x8;

static constexpr int kB  = 4;
static constexpr int kS  = 2048;
static constexpr int kD  = 2048;
static constexpr int kH  = 16;
static constexpr int kDh = 128;
static constexpr int kBS = kB * kS;     // 8192 tokens
static constexpr int kN3 = 3 * kD;      // 6144

DEV __bf16 to_bf16(float f) {
  unsigned u = __float_as_uint(f);
  u += 0x7FFFu + ((u >> 16) & 1u);      // round-to-nearest-even
  unsigned short h = (unsigned short)(u >> 16);
  return __builtin_bit_cast(__bf16, h);
}

// ---- WMMA fragment loaders (contiguous k = 16h + s order, wave32) -------------
// A fragment: 16x32 bf16 tile, row-major with `stride` halves per row.
// lane 0-15: m=lane, K = kb..kb+15 with kb=0
// lane 16-31: m=lane-16, kb=16.
DEV bf16x16 frag_a(const __bf16* tile, int stride, int k0, int lane) {
  int m  = lane & 15;
  int kb = (lane & 16) ? 16 : 0;
  const __bf16* p = tile + m * stride + k0;
  bf16x8 lo = *(const bf16x8*)(p + kb);        // ds_load_b128
  bf16x8 hi = *(const bf16x8*)(p + 8 + kb);    // ds_load_b128
  bf16x16 r;
#pragma unroll
  for (int i = 0; i < 8; ++i) { r[i] = lo[i]; r[i + 8] = hi[i]; }
  return r;
}

// B fragment: 32x16 (KxN) read from an N-major staging tile:
// row n (stride halves), K contiguous. lane 0-15: n=lane, K=0..15;
// lane 16-31: n=lane-16, K=16..31 -> one contiguous 32B read.
DEV bf16x16 frag_b(const __bf16* tileT, int stride, int k0, int lane) {
  int n  = lane & 15;
  int kb = (lane & 16) ? 16 : 0;
  return *(const bf16x16*)(tileT + n * stride + k0 + kb);
}

DEV f32x8 wmma_bf16(bf16x16 a, bf16x16 b, f32x8 c) {
  return __builtin_amdgcn_wmma_f32_16x16x32_bf16(
      /*neg_a=*/false, a, /*neg_b=*/false, b,
      /*c_mod=*/(short)0, c, /*reuse_a=*/false, /*reuse_b=*/false);
}

// ---- Kernel 1: LayerNorm -> bf16 ------------------------------------------
__global__ __launch_bounds__(256)
void ln_kernel(const float* __restrict__ X, const float* __restrict__ w,
               const float* __restrict__ b, __bf16* __restrict__ Xn) {
  const int row = blockIdx.x;
  const float* x = X + (size_t)row * kD;
  __shared__ float red[256];
  float v[8], s = 0.f, s2 = 0.f;
#pragma unroll
  for (int i = 0; i < 8; ++i) {
    v[i] = x[threadIdx.x + i * 256];
    s += v[i]; s2 += v[i] * v[i];
  }
  red[threadIdx.x] = s; __syncthreads();
  for (int o = 128; o; o >>= 1) {
    if (threadIdx.x < o) red[threadIdx.x] += red[threadIdx.x + o];
    __syncthreads();
  }
  float mean = red[0] * (1.f / kD);
  __syncthreads();
  red[threadIdx.x] = s2; __syncthreads();
  for (int o = 128; o; o >>= 1) {
    if (threadIdx.x < o) red[threadIdx.x] += red[threadIdx.x + o];
    __syncthreads();
  }
  float var = red[0] * (1.f / kD) - mean * mean;
  float rstd = rsqrtf(var + 1e-5f);
#pragma unroll
  for (int i = 0; i < 8; ++i) {
    int c = threadIdx.x + i * 256;
    Xn[(size_t)row * kD + c] = to_bf16((v[i] - mean) * rstd * w[c] + b[c]);
  }
}

// ---- Kernel 2: fp32 -> bf16 weight convert --------------------------------
__global__ __launch_bounds__(256)
void cvt_kernel(const float* __restrict__ A, __bf16* __restrict__ O, int n) {
  int i = (blockIdx.x * 256 + threadIdx.x) * 4;
  if (i + 3 < n) {
#pragma unroll
    for (int j = 0; j < 4; ++j) O[i + j] = to_bf16(A[i + j]);
  }
}

// ---- Kernel 3: QKV GEMM (8192x2048 @ 2048x6144) + fused RoPE --------------
// Block tile 128x128, BLK_K=32, 8 waves (4x2 layout), wave tile 32x64.
__global__ __launch_bounds__(256)
void qkv_rope_kernel(const __bf16* __restrict__ Xn, const __bf16* __restrict__ Wi,
                     __bf16* __restrict__ Qh, __bf16* __restrict__ Kh,
                     __bf16* __restrict__ Vh) {
  __shared__ __bf16 As[128 * 32];   // [m][k]
  __shared__ __bf16 Bt[128 * 32];   // [n][k] (transposed stage)
  const int m0 = blockIdx.y * 128;
  const int n0 = blockIdx.x * 128;
  const int t = threadIdx.x, lane = t & 31, wave = t >> 5;
  const int wm = wave >> 1, wn = wave & 1;
  f32x8 acc[2][4] = {};

  for (int k0 = 0; k0 < kD; k0 += 32) {
    __syncthreads();
    { // stage A: 128x32, 8 halves/thread, b128 loads
      int r = t >> 1, c = (t & 1) * 16;
      *(int4*)&As[r * 32 + c] =
          *(const int4*)&Xn[(size_t)(m0 + r) * kD + k0 + c];
      if (k0 + 32 < kD)
        __builtin_prefetch(&Xn[(size_t)(m0 + r) * kD + k0 + 32 + c], 0, 0);
    }
    { // stage B transposed: 32x128 from W_in -> Bt[n][k]
      int kk = t >> 3, ns = (t & 7) * 16;
      const __bf16* g = &Wi[(size_t)(k0 + kk) * kN3 + n0 + ns];
      bf16x8 v0 = *(const bf16x8*)g;
      bf16x8 v1 = *(const bf16x8*)(g + 8);
#pragma unroll
      for (int i = 0; i < 8; ++i) {
        Bt[(ns + i) * 32 + kk]     = v0[i];
        Bt[(ns + 8 + i) * 32 + kk] = v1[i];
      }
    }
    __syncthreads();
    bf16x16 af[2], bfg[4];
#pragma unroll
    for (int i = 0; i < 2; ++i)
      af[i] = frag_a(&As[(wm * 32 + i * 16) * 32], 32, 0, lane);
#pragma unroll
    for (int j = 0; j < 4; ++j)
      bfg[j] = frag_b(&Bt[(wn * 64 + j * 16) * 32], 32, 0, lane);
#pragma unroll
    for (int i = 0; i < 2; ++i)
#pragma unroll
      for (int j = 0; j < 4; ++j)
        acc[i][j] = wmma_bf16(af[i], bfg[j], acc[i][j]);
  }

  // Epilogue: split Q/K/V, apply RoPE (pairs are adjacent lanes), write bf16
  const int cn = lane & 15, hiL = (lane >> 4) & 1;
#pragma unroll
  for (int i = 0; i < 2; ++i) {
#pragma unroll
    for (int j = 0; j < 4; ++j) {
#pragma unroll
      for (int r = 0; r < 8; ++r) {
        int row = m0 + wm * 32 + i * 16 + r + hiL * 8;   // token index
        int col = n0 + wn * 64 + j * 16 + cn;            // 0..6143
        float v = acc[i][j][r];
        float part = __shfl_xor(v, 1, 32);               // RoPE partner
        int which = col >> 11;          // 0:Q 1:K 2:V
        int within = col & 2047;
        int h = within >> 7, d = within & 127;
        int bb = row >> 11, s = row & 2047;
        float ov = v;
        if (which < 2 && d < 64) {      // uniform per sub-tile
          int dpair = d & ~1;
          float freq = __powf(10000.f, -(float)dpair * (1.f / 64.f));
          float ang = (float)s * freq;
          float cs = __cosf(ang), sn = __sinf(ang);
          ov = (d & 1) ? (v * cs + part * sn) : (v * cs - part * sn);
        }
        size_t off = ((size_t)(bb * kH + h) * kS + s) * kDh + d;
        __bf16 o16 = to_bf16(ov);
        if (which == 0)      Qh[off] = o16;
        else if (which == 1) Kh[off] = o16;
        else                 Vh[off] = o16;
      }
    }
  }
}

// ---- Kernel 4: flash attention, one (b,h) x 128 queries per block ---------
__global__ __launch_bounds__(256)
void attn_kernel(const __bf16* __restrict__ Qh, const __bf16* __restrict__ Kh,
                 const __bf16* __restrict__ Vh, __bf16* __restrict__ Ctx) {
  __shared__ __bf16 Qs[128 * 128];      // [q][d]
  __shared__ __bf16 Ks[32 * 128];       // [key][d]  (== B-frag layout for QK^T)
  __shared__ __bf16 Vt[128 * 32];       // [d][key]  (== B-frag layout for PV)
  __shared__ __bf16 Ps[8 * 16 * 32];    // per-wave P tile [m][key]
  const int bh = blockIdx.y;            // b*16 + h
  const int q0 = blockIdx.x * 128;
  const int t = threadIdx.x, lane = t & 31, wave = t >> 5;
  const __bf16* Qg = Qh + (size_t)bh * kS * kDh;
  const __bf16* Kg = Kh + (size_t)bh * kS * kDh;
  const __bf16* Vg = Vh + (size_t)bh * kS * kDh;

  { // stage Q block once: 128x128 halves
    int r = t >> 1, cb = (t & 1) * 64;
#pragma unroll
    for (int i = 0; i < 4; ++i)
      *(int4*)&Qs[r * 128 + cb + i * 16] =
          *(const int4*)&Qg[(size_t)(q0 + r) * kDh + cb + i * 16];
  }

  f32x8 o[8] = {};
  float mrow[8], lsum[8], corrs[8];
#pragma unroll
  for (int r = 0; r < 8; ++r) { mrow[r] = -INFINITY; lsum[r] = 0.f; }
  const float scale = 0.08838834764831845f;   // 1/sqrt(128)
  __bf16* Pw = &Ps[wave * 16 * 32];
  const __bf16* Qw = &Qs[wave * 16 * 128];

  for (int kc = 0; kc < kS; kc += 32) {
    __syncthreads();
    { // stage K (row-major) and V (transposed)
      int kk = t >> 3, dsg = (t & 7) * 16;
      *(int4*)&Ks[kk * 128 + dsg] =
          *(const int4*)&Kg[(size_t)(kc + kk) * kDh + dsg];
      bf16x8 v0 = *(const bf16x8*)&Vg[(size_t)(kc + kk) * kDh + dsg];
      bf16x8 v1 = *(const bf16x8*)&Vg[(size_t)(kc + kk) * kDh + dsg + 8];
#pragma unroll
      for (int i = 0; i < 8; ++i) {
        Vt[(dsg + i) * 32 + kk]     = v0[i];
        Vt[(dsg + 8 + i) * 32 + kk] = v1[i];
      }
    }
    __syncthreads();

    // scores: 16x32 keys = two 16x16 C tiles, Dh=128 -> 4 K-steps
    f32x8 c0 = {}, c1 = {};
#pragma unroll
    for (int dk = 0; dk < 4; ++dk) {
      bf16x16 a  = frag_a(Qw, 128, dk * 32, lane);
      bf16x16 b0 = frag_b(&Ks[0], 128, dk * 32, lane);
      bf16x16 b1 = frag_b(&Ks[16 * 128], 128, dk * 32, lane);
      c0 = wmma_bf16(a, b0, c0);
      c1 = wmma_bf16(a, b1, c1);
    }

    // online softmax, P -> LDS (bf16, A-frag layout)
    const int mloc = (lane >> 4) << 3;
#pragma unroll
    for (int r = 0; r < 8; ++r) {
      float s0 = c0[r] * scale, s1 = c1[r] * scale;
      float mx = fmaxf(s0, s1);
#pragma unroll
      for (int m = 8; m; m >>= 1) mx = fmaxf(mx, __shfl_xor(mx, m, 32));
      float mn = fmaxf(mrow[r], mx);
      corrs[r] = __expf(mrow[r] - mn);
      mrow[r] = mn;
      float p0 = __expf(s0 - mn), p1 = __expf(s1 - mn);
      lsum[r] = lsum[r] * corrs[r] + p0 + p1;
      int mrw = r + mloc;
      Pw[mrw * 32 + (lane & 15)]      = to_bf16(p0);
      Pw[mrw * 32 + 16 + (lane & 15)] = to_bf16(p1);
    }
#pragma unroll
    for (int dt = 0; dt < 8; ++dt)
#pragma unroll
      for (int r = 0; r < 8; ++r) o[dt][r] *= corrs[r];

    asm volatile("" ::: "memory");   // keep P stores before P loads (same wave, DS in-order)
    bf16x16 ap = frag_a(Pw, 32, 0, lane);
#pragma unroll
    for (int dt = 0; dt < 8; ++dt) {
      bf16x16 bv = frag_b(&Vt[dt * 16 * 32], 32, 0, lane);
      o[dt] = wmma_bf16(ap, bv, o[dt]);
    }
  }

  // normalize + write ctx (B,S,D layout, bf16)
  const int bb = bh >> 4, h = bh & 15;
#pragma unroll
  for (int r = 0; r < 8; ++r) {
    float ls = lsum[r];
#pragma unroll
    for (int m = 8; m; m >>= 1) ls += __shfl_xor(ls, m, 32);
    float inv = 1.f / ls;
    int srow = q0 + wave * 16 + r + ((lane >> 4) << 3);
#pragma unroll
    for (int dt = 0; dt < 8; ++dt) {
      int d = dt * 16 + (lane & 15);
      Ctx[((size_t)(bb * kS + srow)) * kD + h * kDh + d] =
          to_bf16(o[dt][r] * inv);
    }
  }
}

// ---- Kernel 5: out GEMM (8192x2048 @ 2048x2048) + residual ----------------
__global__ __launch_bounds__(256)
void out_gemm_kernel(const __bf16* __restrict__ Cx, const __bf16* __restrict__ Wo,
                     const float* __restrict__ X, float* __restrict__ Out) {
  __shared__ __bf16 As[128 * 32];
  __shared__ __bf16 Bt[128 * 32];
  const int m0 = blockIdx.y * 128;
  const int n0 = blockIdx.x * 128;
  const int t = threadIdx.x, lane = t & 31, wave = t >> 5;
  const int wm = wave >> 1, wn = wave & 1;
  f32x8 acc[2][4] = {};

  for (int k0 = 0; k0 < kD; k0 += 32) {
    __syncthreads();
    {
      int r = t >> 1, c = (t & 1) * 16;
      *(int4*)&As[r * 32 + c] =
          *(const int4*)&Cx[(size_t)(m0 + r) * kD + k0 + c];
    }
    {
      int kk = t >> 3, ns = (t & 7) * 16;
      const __bf16* g = &Wo[(size_t)(k0 + kk) * kD + n0 + ns];
      bf16x8 v0 = *(const bf16x8*)g;
      bf16x8 v1 = *(const bf16x8*)(g + 8);
#pragma unroll
      for (int i = 0; i < 8; ++i) {
        Bt[(ns + i) * 32 + kk]     = v0[i];
        Bt[(ns + 8 + i) * 32 + kk] = v1[i];
      }
    }
    __syncthreads();
    bf16x16 af[2], bfg[4];
#pragma unroll
    for (int i = 0; i < 2; ++i)
      af[i] = frag_a(&As[(wm * 32 + i * 16) * 32], 32, 0, lane);
#pragma unroll
    for (int j = 0; j < 4; ++j)
      bfg[j] = frag_b(&Bt[(wn * 64 + j * 16) * 32], 32, 0, lane);
#pragma unroll
    for (int i = 0; i < 2; ++i)
#pragma unroll
      for (int j = 0; j < 4; ++j)
        acc[i][j] = wmma_bf16(af[i], bfg[j], acc[i][j]);
  }
  const int cn = lane & 15, hiL = (lane >> 4) & 1;
#pragma unroll
  for (int i = 0; i < 2; ++i)
#pragma unroll
    for (int j = 0; j < 4; ++j)
#pragma unroll
      for (int r = 0; r < 8; ++r) {
        int row = m0 + wm * 32 + i * 16 + r + hiL * 8;
        int col = n0 + wn * 64 + j * 16 + cn;
        size_t off = (size_t)row * kD + col;
        Out[off] = acc[i][j][r] + X[off];
      }
}

// ---- launcher -------------------------------------------------------------
extern "C" void kernel_launch(void* const* d_in, const int* in_sizes, int n_in,
                              void* d_out, int out_size, void* d_ws, size_t ws_size,
                              hipStream_t stream) {
  (void)in_sizes; (void)n_in; (void)out_size; (void)ws_size;
  const float* X    = (const float*)d_in[0];
  const float* ln_w = (const float*)d_in[1];
  const float* ln_b = (const float*)d_in[2];
  const float* W_in = (const float*)d_in[3];
  const float* W_out= (const float*)d_in[4];
  float* out = (float*)d_out;

  char* ws = (char*)d_ws;
  size_t off = 0;
  __bf16* Xn  = (__bf16*)(ws + off); off += (size_t)kBS * kD  * 2;   // 32 MB
  __bf16* Wi  = (__bf16*)(ws + off); off += (size_t)kD  * kN3 * 2;   // 24 MB
  __bf16* Wo  = (__bf16*)(ws + off); off += (size_t)kD  * kD  * 2;   //  8 MB
  __bf16* Qh  = (__bf16*)(ws + off); off += (size_t)kBS * kD  * 2;   // 32 MB
  __bf16* Kh  = (__bf16*)(ws + off); off += (size_t)kBS * kD  * 2;   // 32 MB
  __bf16* Vh  = (__bf16*)(ws + off); off += (size_t)kBS * kD  * 2;   // 32 MB
  __bf16* Cx  = (__bf16*)(ws + off); off += (size_t)kBS * kD  * 2;   // 32 MB

  ln_kernel<<<kBS, 256, 0, stream>>>(X, ln_w, ln_b, Xn);
  {
    int n1 = kD * kN3;
    cvt_kernel<<<(n1 / 4 + 255) / 256, 256, 0, stream>>>(W_in, Wi, n1);
    int n2 = kD * kD;
    cvt_kernel<<<(n2 / 4 + 255) / 256, 256, 0, stream>>>(W_out, Wo, n2);
  }
  qkv_rope_kernel<<<dim3(kN3 / 128, kBS / 128), 256, 0, stream>>>(Xn, Wi, Qh, Kh, Vh);
  attn_kernel<<<dim3(kS / 128, kB * kH), 256, 0, stream>>>(Qh, Kh, Vh, Cx);
  out_gemm_kernel<<<dim3(kD / 128, kBS / 128), 256, 0, stream>>>(Cx, Wo, X, out);
}
